// LSTM_1271310319989
// MI455X (gfx1250) — matrix operand order repaired and checked by reading, not compiled, not run on hardware
//
#include <hip/hip_runtime.h>
#include <hip/hip_bf16.h>
#include <math.h>

// LSTM: B=32, T=512, D=1024, H=1024
#define BATCH 32
#define SEQT  512
#define DIN   1024
#define HID   1024
#define G4    4096   // 4*H

typedef __attribute__((ext_vector_type(16))) __bf16 v16bf;
typedef __attribute__((ext_vector_type(8)))  __bf16 v8bf;
typedef __attribute__((ext_vector_type(8)))  float  v8f;

static __device__ __forceinline__ v16bf pack16(v8bf lo, v8bf hi) {
    v16bf r;
    #pragma unroll
    for (int i = 0; i < 8; ++i) { r[i] = lo[i]; r[i + 8] = hi[i]; }
    return r;
}

// ---------------- conversion / init kernels ----------------

__global__ __launch_bounds__(256) void lstm_cvt_bf16(const float* __restrict__ src,
                                                     __bf16* __restrict__ dst, int n) {
    int i = blockIdx.x * blockDim.x + threadIdx.x;
    if (i < n) dst[i] = (__bf16)src[i];
}

// src[R][C] (row-major f32) -> dst[C][R] (row-major bf16), i.e. N-major weights
__global__ __launch_bounds__(256) void lstm_cvt_T(const float* __restrict__ src,
                                                  __bf16* __restrict__ dst, int R, int C) {
    int i = blockIdx.x * blockDim.x + threadIdx.x;   // over R*C, c fastest (coalesced read)
    if (i < R * C) {
        int r = i / C;
        int c = i - r * C;
        dst[(size_t)c * R + r] = (__bf16)src[i];
    }
}

__global__ __launch_bounds__(256) void lstm_init_state(__bf16* __restrict__ h0,
                                                       float* __restrict__ c0, int n) {
    int i = blockIdx.x * blockDim.x + threadIdx.x;
    if (i < n) { h0[i] = (__bf16)0.0f; c0[i] = 0.0f; }
}

// ---------------- GEMM1: x_proj[m][g] = x[m][:] . WihT[g][:] + bias[g] ----------------
// grid = (G4/256, (B*T)/32), block = 256 (8 waves).
// Block computes 32 M-rows x 256 N-cols. Each wave: 32x32 (2x2 register-blocked tiles).
// Per K-step: 2 A frags (LDS) + 2 B frags (global) -> 4 WMMAs (1 fragment load / WMMA).
__global__ __launch_bounds__(256) void lstm_gemm_ih(const __bf16* __restrict__ xb,
                                                    const __bf16* __restrict__ wihT,
                                                    const float* __restrict__ bias,
                                                    float* __restrict__ xproj) {
    __shared__ __bf16 Alds[32 * DIN];           // 64 KB: 32 rows of A, full K
    const int tid = threadIdx.x;
    const int m0  = blockIdx.y * 32;

    // cooperative load of A tile (32 x 1024 bf16), 16B per thread per iter
    for (int it = 0; it < 16; ++it) {
        int chunk = it * 256 + tid;             // 4096 chunks of 8 bf16
        int r  = chunk >> 7;                    // 128 chunks per row
        int c8 = (chunk & 127) << 3;
        *(v8bf*)&Alds[r * DIN + c8] = *(const v8bf*)&xb[(size_t)(m0 + r) * DIN + c8];
    }
    __syncthreads();

    const int wave = tid >> 5;
    const int lane = tid & 31;
    const int half = lane >> 4;
    const int lr   = lane & 15;
    const int n0   = blockIdx.x * 256 + wave * 32;
    const int nc0  = n0 + lr;                   // first N tile column
    const int nc1  = n0 + 16 + lr;              // second N tile column
    const __bf16* bp0 = wihT + (size_t)nc0 * DIN;
    const __bf16* bp1 = wihT + (size_t)nc1 * DIN;

    v8f acc00 = {}, acc01 = {}, acc10 = {}, acc11 = {};
    for (int kb = 0; kb < DIN; kb += 32) {
        __builtin_prefetch(bp0 + kb + 128, 0, 3);
        __builtin_prefetch(bp1 + kb + 128, 0, 3);
        // A fragments: K = kb + half*16 + {0..15}, contiguous (matches B)
        v16bf a0 = pack16(*(const v8bf*)&Alds[lr * DIN + kb + half * 16],
                          *(const v8bf*)&Alds[lr * DIN + kb + half * 16 + 8]);
        v16bf a1 = pack16(*(const v8bf*)&Alds[(16 + lr) * DIN + kb + half * 16],
                          *(const v8bf*)&Alds[(16 + lr) * DIN + kb + half * 16 + 8]);
        // B fragments (same k order as A): K = kb + half*16 + {0..15}, contiguous
        v16bf b0 = pack16(*(const v8bf*)&bp0[kb + half * 16],
                          *(const v8bf*)&bp0[kb + half * 16 + 8]);
        v16bf b1 = pack16(*(const v8bf*)&bp1[kb + half * 16],
                          *(const v8bf*)&bp1[kb + half * 16 + 8]);
        acc00 = __builtin_amdgcn_wmma_f32_16x16x32_bf16(false, a0, false, b0,
                                                        (short)0, acc00, false, false);
        acc01 = __builtin_amdgcn_wmma_f32_16x16x32_bf16(false, a0, false, b1,
                                                        (short)0, acc01, false, false);
        acc10 = __builtin_amdgcn_wmma_f32_16x16x32_bf16(false, a1, false, b0,
                                                        (short)0, acc10, false, false);
        acc11 = __builtin_amdgcn_wmma_f32_16x16x32_bf16(false, a1, false, b1,
                                                        (short)0, acc11, false, false);
    }
    // C/D layout: VGPR v -> row (tile_m0 + v + 8*half), col nc{0,1}
    const float bv0 = bias[nc0];
    const float bv1 = bias[nc1];
    #pragma unroll
    for (int v = 0; v < 8; ++v) {
        size_t r0 = (size_t)(m0 + v + 8 * half) * G4;
        size_t r1 = (size_t)(m0 + 16 + v + 8 * half) * G4;
        xproj[r0 + nc0] = acc00[v] + bv0;
        xproj[r0 + nc1] = acc01[v] + bv1;
        xproj[r1 + nc0] = acc10[v] + bv0;
        xproj[r1 + nc1] = acc11[v] + bv1;
    }
}

// ---------------- per-timestep kernel ----------------
// grid = H/16 = 64 blocks, 512 threads (16 waves = 2 K-splits x 2 M-halves x 4 gates).
// gates = xproj[:,t,:] + h_in @ WhhT ; K split halves the serial WMMA chain per step.
__global__ __launch_bounds__(512) void lstm_step(const __bf16* __restrict__ whhT,
                                                 const float* __restrict__ xproj,
                                                 const __bf16* __restrict__ h_in,
                                                 __bf16* __restrict__ h_out,
                                                 float* __restrict__ cst,
                                                 float* __restrict__ out,
                                                 float* __restrict__ hT,
                                                 float* __restrict__ cT,
                                                 int t) {
    __shared__ float glds[2][4][32][16];        // [ksplit][gate][batch][col] = 16 KB
    const int tid  = threadIdx.x;
    const int wave = tid >> 5;
    const int lane = tid & 31;
    const int half = lane >> 4;
    const int lr   = lane & 15;
    const int gate = wave & 3;
    const int mh   = (wave >> 2) & 1;           // 0: batch rows 0-15, 1: rows 16-31
    const int ks   = wave >> 3;                 // K split: 0 -> [0,512), 1 -> [512,1024)
    const int n0   = blockIdx.x * 16;           // hidden-column tile
    const int ncol = gate * HID + n0 + lr;      // global gate column in [0,4H)

    const __bf16* aptr = h_in + (size_t)(mh * 16 + lr) * HID + ks * 512;
    const __bf16* bptr = whhT + (size_t)ncol * HID + ks * 512;

    v8f acc = {};
    for (int kb = 0; kb < 512; kb += 32) {
        __builtin_prefetch(bptr + kb + 128, 0, 3);
        v16bf a = pack16(*(const v8bf*)&aptr[kb + half * 16],
                         *(const v8bf*)&aptr[kb + half * 16 + 8]);
        v16bf b = pack16(*(const v8bf*)&bptr[kb + half * 16],
                         *(const v8bf*)&bptr[kb + half * 16 + 8]);
        acc = __builtin_amdgcn_wmma_f32_16x16x32_bf16(false, a, false, b,
                                                      (short)0, acc, false, false);
    }
    #pragma unroll
    for (int v = 0; v < 8; ++v)
        glds[ks][gate][mh * 16 + v + 8 * half][lr] = acc[v];
    __syncthreads();

    // 32 rows x 16 cols = 512 outputs, 1 per thread
    {
        int e    = tid;
        int row  = e >> 4;                      // batch index
        int col  = e & 15;
        int hidx = n0 + col;                    // hidden index
        size_t m = (size_t)row * SEQT + t;      // xproj row = b*T + t
        const float* xp = xproj + m * G4 + hidx;
        float gi = glds[0][0][row][col] + glds[1][0][row][col] + xp[0 * HID];
        float gf = glds[0][1][row][col] + glds[1][1][row][col] + xp[1 * HID];
        float gg = glds[0][2][row][col] + glds[1][2][row][col] + xp[2 * HID];
        float go = glds[0][3][row][col] + glds[1][3][row][col] + xp[3 * HID];
        float i_t = 1.0f / (1.0f + __expf(-gi));
        float f_t = 1.0f / (1.0f + __expf(-gf));
        float g_t = tanhf(gg);
        float o_t = 1.0f / (1.0f + __expf(-go));
        size_t ci = (size_t)row * HID + hidx;
        float c = f_t * cst[ci] + i_t * g_t;
        float h = o_t * tanhf(c);
        cst[ci]   = c;
        h_out[ci] = (__bf16)h;
        out[(size_t)row * SEQT * HID + (size_t)t * HID + hidx] = h;
        hT[ci] = h;                             // last step's value survives
        cT[ci] = c;
    }
}

// ---------------- host launch ----------------

extern "C" void kernel_launch(void* const* d_in, const int* in_sizes, int n_in,
                              void* d_out, int out_size, void* d_ws, size_t ws_size,
                              hipStream_t stream) {
    const float* x    = (const float*)d_in[0];   // [B,T,D]
    const float* wih  = (const float*)d_in[1];   // [D,4H]
    const float* whh  = (const float*)d_in[2];   // [H,4H]
    const float* bias = (const float*)d_in[3];   // [4H]
    float* out = (float*)d_out;                  // [B,T,H] ++ [B,H] ++ [B,H]

    char* ws = (char*)d_ws;
    size_t off = 0;
    __bf16* xb    = (__bf16*)(ws + off); off += (size_t)BATCH * SEQT * DIN * 2;   // 32 MB
    __bf16* wihT  = (__bf16*)(ws + off); off += (size_t)G4 * DIN * 2;             // 8 MB
    __bf16* whhT  = (__bf16*)(ws + off); off += (size_t)G4 * HID * 2;             // 8 MB
    float*  xproj = (float*)(ws + off);  off += (size_t)BATCH * SEQT * G4 * 4;    // 256 MB
    float*  cst   = (float*)(ws + off);  off += (size_t)BATCH * HID * 4;          // 128 KB
    __bf16* hb0   = (__bf16*)(ws + off); off += (size_t)BATCH * HID * 2;          // 64 KB
    __bf16* hb1   = (__bf16*)(ws + off); off += (size_t)BATCH * HID * 2;          // 64 KB

    float* hT = out + (size_t)BATCH * SEQT * HID;
    float* cT = hT + (size_t)BATCH * HID;

    // 1) convert x to bf16
    {
        int n = BATCH * SEQT * DIN;
        lstm_cvt_bf16<<<(n + 255) / 256, 256, 0, stream>>>(x, xb, n);
    }
    // 2) transpose-convert weights to N-major bf16
    {
        int n = DIN * G4;
        lstm_cvt_T<<<(n + 255) / 256, 256, 0, stream>>>(wih, wihT, DIN, G4);
        lstm_cvt_T<<<(n + 255) / 256, 256, 0, stream>>>(whh, whhT, HID, G4);
    }
    // 3) zero h0 / c0
    {
        int n = BATCH * HID;
        lstm_init_state<<<(n + 255) / 256, 256, 0, stream>>>(hb0, cst, n);
    }
    // 4) big input-projection GEMM: [16384 x 1024] x [1024 x 4096]
    {
        dim3 grid(G4 / 256, (BATCH * SEQT) / 32);
        lstm_gemm_ih<<<grid, 256, 0, stream>>>(xb, wihT, bias, xproj);
    }
    // 5) recurrence: one launch per timestep, double-buffered h
    for (int t = 0; t < SEQT; ++t) {
        const __bf16* hin  = (t & 1) ? hb1 : hb0;
        __bf16*       hout = (t & 1) ? hb0 : hb1;
        lstm_step<<<HID / 16, 512, 0, stream>>>(whhT, xproj, hin, hout, cst,
                                                out, hT, cT, t);
    }
}
